// MultiHeadAttention_47510928228581
// MI455X (gfx1250) — matrix operand order repaired and checked by reading, not compiled, not run on hardware
//
#include <hip/hip_runtime.h>

// Problem constants (match reference)
#define B_   2
#define T_   2048
#define E_   1024
#define H_   16
#define DH_  64
#define M_   (B_ * T_)   // 4096 rows of x / q / k / v

typedef _Float16 half8  __attribute__((ext_vector_type(8)));
typedef _Float16 half16 __attribute__((ext_vector_type(16)));
typedef float    float8 __attribute__((ext_vector_type(8)));
typedef unsigned int uint4v __attribute__((ext_vector_type(4)));
typedef int          int4v  __attribute__((ext_vector_type(4)));
typedef int          int8v  __attribute__((ext_vector_type(8)));

// LDS row strides (halves) for TDM-staged tiles
#define LROW 72   // 64 halves (128B) + 16B pad  -> pad_interval=32dw (code 4)
#define VROW 40   // 32 halves (64B)  + 16B pad  -> pad_interval=16dw (code 3)

// ---------------------------------------------------------------------------
// WMMA fragment helpers (wave32, V_WMMA_F32_16X16X32_F16)
// A (16x32, row-major): lane L = row m=L&15; half hs=L>>4 holds
//   K = hs*16..hs*16+15 contiguously -> 16B loads at +0 and +8 (in halves).
// B (32x16) from Bt row-major [N][K]: lane L = col n=L&15; half hs holds
//   K = hs*16..hs*16+15 contiguously -> 16B loads at +0 and +8.
// C/D (16x16 f32): VGPR r of lane L = C[r + 8*(L>>4)][L&15].
// ---------------------------------------------------------------------------
static __device__ __forceinline__ half16 ld_fragA(const _Float16* pk) {
  union { half16 v; half8 h[2]; } u;
  u.h[0] = *(const half8*)(pk);
  u.h[1] = *(const half8*)(pk + 8);
  return u.v;
}
static __device__ __forceinline__ half16 ld_fragB(const _Float16* pk) {
  union { half16 v; half8 h[2]; } u;
  u.h[0] = *(const half8*)(pk);
  u.h[1] = *(const half8*)(pk + 8);
  return u.v;
}
static __device__ __forceinline__ float8 wmma16(half16 a, half16 b, float8 c) {
  return __builtin_amdgcn_wmma_f32_16x16x32_f16(false, a, false, b, (short)0, c,
                                                false, false);
}

// ---------------------------------------------------------------------------
// Tensor Data Mover: async-load a 2D f16 tile [rows x tile_d0 halves] from a
// row-major matrix (row stride = stride0 elements) into LDS, with LDS row
// padding (pad 4 dwords every (1<<pad_code)*2 dwords). Tracked by TENSORcnt.
// ---------------------------------------------------------------------------
static __device__ __forceinline__ unsigned lds_off_of(const void* p) {
  return (unsigned)(unsigned long long)p;   // flat LDS aperture: low 32b = offset
}

static __device__ __forceinline__ void tdm_load_2d(const _Float16* gsrc,
                                                   unsigned lds_byte_off,
                                                   unsigned tile_d0,
                                                   unsigned rows,
                                                   unsigned stride0,
                                                   unsigned pad_code) {
  unsigned long long ga = (unsigned long long)(size_t)gsrc;
  uint4v g0;
  g0[0] = 1u;                                            // count=1, user mode
  g0[1] = lds_byte_off;                                  // lds_addr
  g0[2] = (unsigned)(ga & 0xFFFFFFFFu);                  // global_addr[31:0]
  g0[3] = (unsigned)((ga >> 32) & 0x01FFFFFFu) | (2u << 30); // ga[56:32] | type=2
  int8v g1;
  // data_size=1 (2B), pad_enable, pad_interval=pad_code, pad_amount=3 (4 dwords)
  g1[0] = (int)((1u << 16) | (1u << 20) | (pad_code << 22) | (3u << 25));
  g1[1] = (int)((stride0 & 0xFFFFu) << 16);              // tensor_dim0[15:0]
  g1[2] = (int)(((stride0 >> 16) & 0xFFFFu) | ((rows & 0xFFFFu) << 16)); // td0 hi|td1 lo
  g1[3] = (int)(((rows >> 16) & 0xFFFFu) | (tile_d0 << 16)); // td1 hi | tile_dim0
  g1[4] = (int)rows;                                     // tile_dim1 (tile_dim2=0)
  g1[5] = (int)stride0;                                  // tensor_dim0_stride lo32
  g1[6] = 0;
  g1[7] = 0;
  int4v gz = {0, 0, 0, 0};
#if defined(__clang_major__) && (__clang_major__ >= 23)
  int8v gz8 = {0, 0, 0, 0, 0, 0, 0, 0};
  __builtin_amdgcn_tensor_load_to_lds(g0, g1, gz, gz, gz8, 0);
#else
  __builtin_amdgcn_tensor_load_to_lds(g0, g1, gz, gz, 0);
#endif
}

// ---------------------------------------------------------------------------
// Kernel 1: fp32 -> f16 elementwise convert
// ---------------------------------------------------------------------------
__global__ __launch_bounds__(256) void cvt_f16_kernel(
    const float* __restrict__ in, _Float16* __restrict__ outp, int n) {
  int i = blockIdx.x * 256 + threadIdx.x;
  if (i < n) outp[i] = (_Float16)in[i];
}

// ---------------------------------------------------------------------------
// Kernel 2: 1024x1024 fp32 [K][N] -> f16 transposed [N][K]
// ---------------------------------------------------------------------------
__global__ __launch_bounds__(256) void transpose_cvt_kernel(
    const float* __restrict__ in, _Float16* __restrict__ outp) {
  __shared__ float tile[32][33];
  const int tx = threadIdx.x, ty = threadIdx.y;
  const int x  = blockIdx.x * 32 + tx;
  const int y0 = blockIdx.y * 32;
#pragma unroll
  for (int j = 0; j < 32; j += 8)
    tile[ty + j][tx] = in[(size_t)(y0 + ty + j) * 1024 + x];
  __syncthreads();
  const int ox  = blockIdx.y * 32 + tx;
  const int oy0 = blockIdx.x * 32;
#pragma unroll
  for (int j = 0; j < 32; j += 8)
    outp[(size_t)(oy0 + ty + j) * 1024 + ox] = (_Float16)tile[tx][ty + j];
}

// ---------------------------------------------------------------------------
// Kernel 3: fused QKV projection, TDM-staged double-buffered pipeline.
// Block = 256 threads (8 waves) -> 128x64 output tile; wave = 16x64.
// z==0 -> Q (row-major), z==1 -> K (row-major),
// z==2 -> V stored TRANSPOSED per head: Vt[((b*H + h)*DH + d)*T + t]
// ---------------------------------------------------------------------------
__global__ __launch_bounds__(256) void gemm_qkv_kernel(
    const _Float16* __restrict__ A,
    const _Float16* __restrict__ Bt0, const _Float16* __restrict__ Bt1,
    const _Float16* __restrict__ Bt2,
    _Float16* __restrict__ C0, _Float16* __restrict__ C1,
    _Float16* __restrict__ Vt) {
  __shared__ _Float16 sA[2][128 * LROW];
  __shared__ _Float16 sB[2][64 * LROW];
  const _Float16* Bt = (blockIdx.z == 0) ? Bt0 : ((blockIdx.z == 1) ? Bt1 : Bt2);
  const int lane = threadIdx.x & 31;
  const int wave = threadIdx.x >> 5;
  const int ln = lane & 15;
  const int hs = lane >> 4;
  const int mblk = blockIdx.x * 128;
  const int m0 = mblk + wave * 16;
  const int n0 = blockIdx.y * 64;

  const _Float16* ga = A  + (size_t)mblk * E_;
  const _Float16* gb = Bt + (size_t)n0 * E_;

  if (wave == 0) {
    tdm_load_2d(ga, lds_off_of(&sA[0][0]), 64, 128, 1024, 4);
    tdm_load_2d(gb, lds_off_of(&sB[0][0]), 64, 64, 1024, 4);
  }

  float8 acc[4] = {};
  int buf = 0;
  for (int kb = 0; kb < E_; kb += 64) {
    if (wave == 0) {
      if (kb + 64 < E_) {
        tdm_load_2d(ga + kb + 64, lds_off_of(&sA[buf ^ 1][0]), 64, 128, 1024, 4);
        tdm_load_2d(gb + kb + 64, lds_off_of(&sB[buf ^ 1][0]), 64, 64, 1024, 4);
        __builtin_amdgcn_s_wait_tensorcnt(2);  // current tile's pair complete
      } else {
        __builtin_amdgcn_s_wait_tensorcnt(0);
      }
    }
    __syncthreads();
    const _Float16* ap = &sA[buf][0] + (wave * 16 + ln) * LROW + hs * 16;
    const _Float16* bp = &sB[buf][0] + (size_t)ln * LROW + hs * 16;
#pragma unroll
    for (int s = 0; s < 2; ++s) {
      const half16 a = ld_fragA(ap + s * 32);
#pragma unroll
      for (int t = 0; t < 4; ++t) {
        const half16 bf = ld_fragB(bp + t * 16 * LROW + s * 32);
        acc[t] = wmma16(a, bf, acc[t]);
      }
    }
    __syncthreads();
    buf ^= 1;
  }

  if (blockIdx.z == 2) {
    // V: store transposed per head -> Vt[((b*H + h)*DH + d)*T + t]
#pragma unroll
    for (int t = 0; t < 4; ++t)
#pragma unroll
      for (int r = 0; r < 8; ++r) {
        const int m = m0 + r + 8 * hs;              // b*T + tq
        const int bb = m >> 11;                     // / T_
        const int tq = m & (T_ - 1);
        const int n = n0 + t * 16 + ln;             // h*DH + d
        Vt[(((size_t)bb * H_) * DH_ + n) * T_ + tq] = (_Float16)acc[t][r];
      }
  } else {
    _Float16* C = (blockIdx.z == 0) ? C0 : C1;
#pragma unroll
    for (int t = 0; t < 4; ++t)
#pragma unroll
      for (int r = 0; r < 8; ++r)
        C[(size_t)(m0 + r + 8 * hs) * E_ + n0 + t * 16 + ln] = (_Float16)acc[t][r];
  }
}

// ---------------------------------------------------------------------------
// Kernel 4: causal flash attention. Q row-major, K row-major, V transposed.
// Block = (b, h, 64 q rows); 4 waves x 16 q rows; 32-key chunks with online
// softmax. K (32x64) and V^T (64x32) tiles are TDM double-buffered in LDS.
// V^T fragments are pulled into registers right after the publish barrier so
// the post-softmax barrier also fences buffer reuse for the next TDM issue.
// ---------------------------------------------------------------------------
__global__ __launch_bounds__(128) void attn_kernel(
    const _Float16* __restrict__ Qh, const _Float16* __restrict__ Kh,
    const _Float16* __restrict__ Vt, _Float16* __restrict__ Yh) {
  __shared__ _Float16 sK[2][32 * LROW];
  __shared__ _Float16 sV[2][64 * VROW];
  __shared__ _Float16 sPb[4 * 16 * 40];
  const int lane = threadIdx.x & 31;
  const int wave = threadIdx.x >> 5;
  const int ln = lane & 15;
  const int hs = lane >> 4;
  const int b = blockIdx.z, h = blockIdx.y;
  const int qblk  = blockIdx.x << 6;
  const int qbase = qblk + wave * 16;

  const _Float16* Qp  = Qh + (size_t)(b * T_ + qbase) * E_ + h * DH_;
  const _Float16* Kp  = Kh + (size_t)(b * T_) * E_ + h * DH_;
  const _Float16* Vtp = Vt + ((size_t)(b * H_ + h)) * DH_ * T_;

  const half16 qf0 = ld_fragA(Qp + (size_t)ln * E_ + hs * 16);
  const half16 qf1 = ld_fragA(Qp + (size_t)ln * E_ + 32 + hs * 16);

  float8 acc[4] = {};
  float m_r[8], l_r[8];
#pragma unroll
  for (int r = 0; r < 8; ++r) { m_r[r] = -__builtin_inff(); l_r[r] = 0.0f; }

  const float scale = 0.125f;
  const int nch = (qblk >> 5) + 2;   // uniform across the block (causal)
  _Float16* Pb = sPb + wave * (16 * 40);

  if (wave == 0) {
    tdm_load_2d(Kp,  lds_off_of(&sK[0][0]), 64, 32, 1024, 4);
    tdm_load_2d(Vtp, lds_off_of(&sV[0][0]), 32, 64, 2048, 3);
  }

  int buf = 0;
  for (int ch = 0; ch < nch; ++ch) {
    const int k0 = ch << 5;
    if (wave == 0) {
      if (ch + 1 < nch) {
        tdm_load_2d(Kp + (size_t)(k0 + 32) * E_, lds_off_of(&sK[buf ^ 1][0]),
                    64, 32, 1024, 4);
        tdm_load_2d(Vtp + k0 + 32, lds_off_of(&sV[buf ^ 1][0]), 32, 64, 2048, 3);
        __builtin_amdgcn_s_wait_tensorcnt(2);
      } else {
        __builtin_amdgcn_s_wait_tensorcnt(0);
      }
    }
    __syncthreads();                    // publish current K/V tiles

    // V^T fragments into registers now (frees the buffer by the next barrier)
    half16 vf[4];
#pragma unroll
    for (int t = 0; t < 4; ++t)
      vf[t] = ld_fragB(&sV[buf][0] + (t * 16 + ln) * VROW + hs * 16);

    // S = Q K^T for keys [k0, k0+32): two 16x16 f32 tiles from LDS K tile
    float8 s0 = {}, s1 = {};
    const _Float16* kb = &sK[buf][0] + (size_t)ln * LROW + hs * 16;
    s0 = wmma16(qf0, ld_fragB(kb), s0);
    s0 = wmma16(qf1, ld_fragB(kb + 32), s0);
    s1 = wmma16(qf0, ld_fragB(kb + 16 * LROW), s1);
    s1 = wmma16(qf1, ld_fragB(kb + 16 * LROW + 32), s1);

    // causal mask + online softmax (row reductions across 16 lanes)
    float p0[8], p1[8], alpha[8];
#pragma unroll
    for (int r = 0; r < 8; ++r) {
      const int q = qbase + r + 8 * hs;
      float v0 = (k0 + ln      <= q) ? s0[r] * scale : -__builtin_inff();
      float v1 = (k0 + 16 + ln <= q) ? s1[r] * scale : -__builtin_inff();
      float rm = fmaxf(v0, v1);
      rm = fmaxf(rm, __shfl_xor(rm, 1, 32));
      rm = fmaxf(rm, __shfl_xor(rm, 2, 32));
      rm = fmaxf(rm, __shfl_xor(rm, 4, 32));
      rm = fmaxf(rm, __shfl_xor(rm, 8, 32));
      const float mn = fmaxf(m_r[r], rm);
      alpha[r] = __expf(m_r[r] - mn);
      m_r[r] = mn;
      p0[r] = __expf(v0 - mn);
      p1[r] = __expf(v1 - mn);
      float rs = p0[r] + p1[r];
      rs += __shfl_xor(rs, 1, 32);
      rs += __shfl_xor(rs, 2, 32);
      rs += __shfl_xor(rs, 4, 32);
      rs += __shfl_xor(rs, 8, 32);
      l_r[r] = alpha[r] * l_r[r] + rs;
    }
#pragma unroll
    for (int t = 0; t < 4; ++t)
#pragma unroll
      for (int r = 0; r < 8; ++r) acc[t][r] *= alpha[r];

    // P (C-layout) -> per-wave LDS for A-fragment relayout
#pragma unroll
    for (int r = 0; r < 8; ++r) {
      const int row = r + 8 * hs;
      Pb[row * 40 + ln]      = (_Float16)p0[r];
      Pb[row * 40 + 16 + ln] = (_Float16)p1[r];
    }
    __syncthreads();                    // P ordering + buffer-reuse fence

    const half16 pf = ld_fragA(Pb + ln * 40 + hs * 16);
#pragma unroll
    for (int t = 0; t < 4; ++t) acc[t] = wmma16(pf, vf[t], acc[t]);
    buf ^= 1;
  }

#pragma unroll
  for (int t = 0; t < 4; ++t)
#pragma unroll
    for (int r = 0; r < 8; ++r) {
      const int q = qbase + r + 8 * hs;
      Yh[(size_t)(b * T_ + q) * E_ + h * DH_ + t * 16 + ln] =
          (_Float16)(acc[t][r] / l_r[r]);
    }
}

// ---------------------------------------------------------------------------
// Kernel 5: output projection out = Y x Wo + bo (fp32 out), TDM pipeline.
// ---------------------------------------------------------------------------
__global__ __launch_bounds__(256) void gemm_out_kernel(
    const _Float16* __restrict__ A, const _Float16* __restrict__ Bt,
    const float* __restrict__ bias, float* __restrict__ outp) {
  __shared__ _Float16 sA[2][128 * LROW];
  __shared__ _Float16 sB[2][64 * LROW];
  const int lane = threadIdx.x & 31;
  const int wave = threadIdx.x >> 5;
  const int ln = lane & 15;
  const int hs = lane >> 4;
  const int mblk = blockIdx.x * 128;
  const int m0 = mblk + wave * 16;
  const int n0 = blockIdx.y * 64;

  const _Float16* ga = A  + (size_t)mblk * E_;
  const _Float16* gb = Bt + (size_t)n0 * E_;

  if (wave == 0) {
    tdm_load_2d(ga, lds_off_of(&sA[0][0]), 64, 128, 1024, 4);
    tdm_load_2d(gb, lds_off_of(&sB[0][0]), 64, 64, 1024, 4);
  }

  float8 acc[4] = {};
  int buf = 0;
  for (int kb = 0; kb < E_; kb += 64) {
    if (wave == 0) {
      if (kb + 64 < E_) {
        tdm_load_2d(ga + kb + 64, lds_off_of(&sA[buf ^ 1][0]), 64, 128, 1024, 4);
        tdm_load_2d(gb + kb + 64, lds_off_of(&sB[buf ^ 1][0]), 64, 64, 1024, 4);
        __builtin_amdgcn_s_wait_tensorcnt(2);
      } else {
        __builtin_amdgcn_s_wait_tensorcnt(0);
      }
    }
    __syncthreads();
    const _Float16* ap = &sA[buf][0] + (wave * 16 + ln) * LROW + hs * 16;
    const _Float16* bp = &sB[buf][0] + (size_t)ln * LROW + hs * 16;
#pragma unroll
    for (int s = 0; s < 2; ++s) {
      const half16 a = ld_fragA(ap + s * 32);
#pragma unroll
      for (int t = 0; t < 4; ++t) {
        const half16 bf = ld_fragB(bp + t * 16 * LROW + s * 32);
        acc[t] = wmma16(a, bf, acc[t]);
      }
    }
    __syncthreads();
    buf ^= 1;
  }
#pragma unroll
  for (int t = 0; t < 4; ++t)
#pragma unroll
    for (int r = 0; r < 8; ++r) {
      const int col = n0 + t * 16 + ln;
      outp[(size_t)(m0 + r + 8 * hs) * E_ + col] = acc[t][r] + bias[col];
    }
}

// ---------------------------------------------------------------------------
// Launcher
// ---------------------------------------------------------------------------
extern "C" void kernel_launch(void* const* d_in, const int* in_sizes, int n_in,
                              void* d_out, int out_size, void* d_ws,
                              size_t ws_size, hipStream_t stream) {
  (void)in_sizes; (void)n_in; (void)out_size; (void)ws_size;
  const float* x  = (const float*)d_in[0];
  const float* Wq = (const float*)d_in[1];
  const float* Wk = (const float*)d_in[2];
  const float* Wv = (const float*)d_in[3];
  const float* Wo = (const float*)d_in[4];
  const float* bo = (const float*)d_in[5];
  float* out = (float*)d_out;

  char* ws = (char*)d_ws;
  size_t off = 0;
  _Float16* xh  = (_Float16*)(ws + off); off += (size_t)M_ * E_ * 2;
  _Float16* WqT = (_Float16*)(ws + off); off += (size_t)E_ * E_ * 2;
  _Float16* WkT = (_Float16*)(ws + off); off += (size_t)E_ * E_ * 2;
  _Float16* WvT = (_Float16*)(ws + off); off += (size_t)E_ * E_ * 2;
  _Float16* WoT = (_Float16*)(ws + off); off += (size_t)E_ * E_ * 2;
  _Float16* Qh  = (_Float16*)(ws + off); off += (size_t)M_ * E_ * 2;
  _Float16* Kh  = (_Float16*)(ws + off); off += (size_t)M_ * E_ * 2;
  _Float16* Vt  = (_Float16*)(ws + off); off += (size_t)M_ * E_ * 2;  // transposed V
  _Float16* Yh  = (_Float16*)(ws + off); off += (size_t)M_ * E_ * 2;

  cvt_f16_kernel<<<(M_ * E_) / 256, 256, 0, stream>>>(x, xh, M_ * E_);
  dim3 tg(32, 32), tb(32, 8);
  transpose_cvt_kernel<<<tg, tb, 0, stream>>>(Wq, WqT);
  transpose_cvt_kernel<<<tg, tb, 0, stream>>>(Wk, WkT);
  transpose_cvt_kernel<<<tg, tb, 0, stream>>>(Wv, WvT);
  transpose_cvt_kernel<<<tg, tb, 0, stream>>>(Wo, WoT);
  gemm_qkv_kernel<<<dim3(M_ / 128, E_ / 64, 3), 256, 0, stream>>>(
      xh, WqT, WkT, WvT, Qh, Kh, Vt);
  attn_kernel<<<dim3(T_ / 64, H_, B_), 128, 0, stream>>>(Qh, Kh, Vt, Yh);
  gemm_out_kernel<<<dim3(M_ / 128, E_ / 64), 256, 0, stream>>>(Yh, WoT, bo, out);
}
